// SlidingWindowAttention_3685081940650
// MI455X (gfx1250) — compile-verified, matrix operand order checked by reading, not run on hardware
//
#include <hip/hip_runtime.h>
#include <hip/hip_bf16.h>

typedef __bf16 bf16_t;
typedef __attribute__((ext_vector_type(16))) __bf16 v16bf;
typedef __attribute__((ext_vector_type(8)))  __bf16 v8bf;
typedef __attribute__((ext_vector_type(4)))  __bf16 v4bf;
typedef __attribute__((ext_vector_type(8)))  float  v8f;

#define DEVINL __device__ __forceinline__

static constexpr int kE   = 1024;   // embed dim
static constexpr int kH   = 16;     // heads
static constexpr int kD   = 64;     // head dim
static constexpr int kWin = 128;    // window
static constexpr int kB   = 2;      // batch
static constexpr int kS   = 2048;   // seq len
static constexpr int kM   = kB * kS; // 4096 rows for the GEMMs

// ---------------------------------------------------------------------------
// CDNA5 async global->LDS copy (16B per lane), tracked by ASYNCcnt.
// LDS destination offset taken from a VGPR; global addr from 64-bit VGPR pair.
// ---------------------------------------------------------------------------
DEVINL void async_copy_b128(const void* gptr, void* lds_ptr) {
  const unsigned lds = (unsigned)(size_t)lds_ptr;           // low 32b = LDS offset
  const unsigned long long ga = (unsigned long long)(size_t)gptr;
  asm volatile("global_load_async_to_lds_b128 %0, %1, off"
               :: "v"(lds), "v"(ga) : "memory");
}
DEVINL void async_wait0() {
  asm volatile("s_wait_asynccnt 0x0" ::: "memory");
}

// ---------------------------------------------------------------------------
// Fragment loader for v_wmma_f32_16x16x32_bf16.
// 16-bit A-matrix layout (ISA 7.12.2): lane L(0..15) holds row M=L,
//   halves 0..7 = K[ko..ko+7], halves 8..15 = K[ko+16..ko+23], ko=0;
// lanes 16..31 hold the same rows with ko=8. B mirrors with N as row role.
// ---------------------------------------------------------------------------
DEVINL v16bf load_frag(const bf16_t* __restrict__ base, int stride, int kbase, int lane) {
  const int r  = lane & 15;
  const int ko = kbase + ((lane >> 4) << 3);
  const bf16_t* p = base + (size_t)r * (size_t)stride + ko;
  v8bf lo = *reinterpret_cast<const v8bf*>(p);
  v8bf hi = *reinterpret_cast<const v8bf*>(p + 16);
  return __builtin_shufflevector(lo, hi, 0,1,2,3,4,5,6,7,8,9,10,11,12,13,14,15);
}

DEVINL v8f wmma_bf16(v16bf a, v16bf b, v8f c) {
  return __builtin_amdgcn_wmma_f32_16x16x32_bf16(false, a, false, b, (short)0, c, false, false);
}

// ---------------------------------------------------------------------------
// C = A[M,K] * W[N,K]^T + bias[N]. Block tile 128x128, K-step 32, 8 waves.
// fp32 A is converted to bf16 through VGPRs; bf16 A is async-copied to LDS.
// ---------------------------------------------------------------------------
template <typename TIN, typename TOUT>
__global__ __launch_bounds__(256) void gemm_bias_kernel(
    const TIN* __restrict__ A, const float* __restrict__ W,
    const float* __restrict__ bias, TOUT* __restrict__ C,
    int M, int N, int K)
{
  constexpr int BM = 128, BN = 128, BK = 32, STR = BK + 8;
  __shared__ bf16_t As[BM * STR];
  __shared__ bf16_t Bs[BN * STR];

  const int tid  = threadIdx.x;
  const int lane = tid & 31;
  const int wave = tid >> 5;
  const int bm   = blockIdx.y * BM;
  const int bn   = blockIdx.x * BN;
  const int wr   = (wave >> 1) * 32;
  const int wc   = (wave & 1) * 64;

  v8f acc[2][4] = {};

  for (int k0 = 0; k0 < K; k0 += BK) {
    // ---- stage W slab (always fp32 -> bf16), and A slab if fp32 ----
#pragma unroll
    for (int i = 0; i < 4; ++i) {
      const int slot = tid + 256 * i;   // 0..1023
      const int row  = slot >> 3;       // 0..127
      const int c4   = (slot & 7) << 2; // 0,4,...,28
      const float4 wv = *reinterpret_cast<const float4*>(W + (size_t)(bn + row) * K + k0 + c4);
      v4bf sb;
      sb[0] = (bf16_t)wv.x; sb[1] = (bf16_t)wv.y; sb[2] = (bf16_t)wv.z; sb[3] = (bf16_t)wv.w;
      *reinterpret_cast<v4bf*>(&Bs[row * STR + c4]) = sb;
      if constexpr (sizeof(TIN) == 4) {
        const float4 av = *reinterpret_cast<const float4*>(A + (size_t)(bm + row) * K + k0 + c4);
        v4bf sa;
        sa[0] = (bf16_t)av.x; sa[1] = (bf16_t)av.y; sa[2] = (bf16_t)av.z; sa[3] = (bf16_t)av.w;
        *reinterpret_cast<v4bf*>(&As[row * STR + c4]) = sa;
      }
    }
    if constexpr (sizeof(TIN) == 2) {
      // bf16 A slab: async DMA straight into LDS, no VGPR round-trip
#pragma unroll
      for (int i = 0; i < 2; ++i) {
        const int idx = tid + 256 * i;  // 0..511 chunks of 8 elems (16B)
        const int row = idx >> 2;       // 0..127
        const int c8  = (idx & 3) << 3; // 0,8,16,24
        async_copy_b128(A + (size_t)(bm + row) * K + k0 + c8, &As[row * STR + c8]);
      }
      async_wait0();
    }
    if (k0 + BK < K) {
      __builtin_prefetch(A + (size_t)(bm + (tid >> 3)) * K + k0 + BK, 0, 1);
      __builtin_prefetch(W + (size_t)(bn + (tid >> 3)) * K + k0 + BK, 0, 1);
    }
    __syncthreads();

    v16bf af[2], bfr[4];
#pragma unroll
    for (int i = 0; i < 2; ++i) af[i]  = load_frag(&As[(wr + i * 16) * STR], STR, 0, lane);
#pragma unroll
    for (int j = 0; j < 4; ++j) bfr[j] = load_frag(&Bs[(wc + j * 16) * STR], STR, 0, lane);
#pragma unroll
    for (int i = 0; i < 2; ++i)
#pragma unroll
      for (int j = 0; j < 4; ++j)
        acc[i][j] = wmma_bf16(af[i], bfr[j], acc[i][j]);
    __syncthreads();
  }

  const int mrow0 = bm + wr + ((lane >> 4) << 3);
  const int ncol0 = bn + wc + (lane & 15);
#pragma unroll
  for (int i = 0; i < 2; ++i) {
#pragma unroll
    for (int j = 0; j < 4; ++j) {
      const int col = ncol0 + j * 16;
      const float bv = bias[col];
#pragma unroll
      for (int vg = 0; vg < 8; ++vg) {
        const int row = mrow0 + i * 16 + vg;
        const float val = acc[i][j][vg] + bv;
        if constexpr (sizeof(TOUT) == 4) C[(size_t)row * N + col] = val;
        else                             C[(size_t)row * N + col] = (TOUT)val;
      }
    }
  }
}

// ---------------------------------------------------------------------------
// Sliding-window attention over bf16 Q/K/V stored [B,S,H*D].
// Grid: (S/128, H, B); 8 waves, 16 query rows each. Window 128 => key tiles
// {qt-1, qt, qt+1}; online softmax; QK^T and PV both via WMMA.
// K tile is async-DMA'd to LDS once per block (was 8x-duplicated global
// loads); the K buffer is reused for the per-wave P tiles (phase-disjoint).
// ---------------------------------------------------------------------------
__global__ __launch_bounds__(256) void swa_kernel(
    const bf16_t* __restrict__ Q, const bf16_t* __restrict__ K,
    const bf16_t* __restrict__ V, bf16_t* __restrict__ O)
{
  constexpr int PSTR = 136;               // 128+pad, 16B-aligned rows
  __shared__ bf16_t Vt[kD * PSTR];        // V^T tile: [d][key]
  __shared__ bf16_t KP[128 * PSTR];       // phase 1: K tile [key][d]; phase 2: P tiles

  const int tid  = threadIdx.x;
  const int lane = tid & 31;
  const int wave = tid >> 5;
  const int qt   = blockIdx.x;
  const int h    = blockIdx.y;
  const int b    = blockIdx.z;

  const size_t hbase = ((size_t)b * kS) * kE + (size_t)h * kD;
  const int q0 = qt * 128;
  const int wq = q0 + wave * 16;

  const v16bf qa0 = load_frag(Q + hbase + (size_t)wq * kE, kE, 0,  lane);
  const v16bf qa1 = load_frag(Q + hbase + (size_t)wq * kE, kE, 32, lane);

  float mrow[8], lrow[8];
  v8f o[4] = {};
#pragma unroll
  for (int vg = 0; vg < 8; ++vg) { mrow[vg] = -__builtin_inff(); lrow[vg] = 0.f; }

  const int qrow_base = wq + ((lane >> 4) << 3);
  const int ncol = lane & 15;

  for (int kt = qt - 1; kt <= qt + 1; ++kt) {
    if (kt < 0 || kt >= kS / 128) continue;
    const int k0 = kt * 128;

    // ---- phase 0: stage K (async DMA, row-major) and V^T (transpose) ----
#pragma unroll
    for (int it = 0; it < 4; ++it) {
      const int idx = tid + 256 * it;     // 0..1023 chunks of 8 elems
      const int key = idx >> 3;           // 0..127
      const int c8  = (idx & 7) << 3;     // 0..56
      async_copy_b128(K + hbase + (size_t)(k0 + key) * kE + c8, &KP[key * PSTR + c8]);
      const v8bf vv = *reinterpret_cast<const v8bf*>(V + hbase + (size_t)(k0 + key) * kE + c8);
#pragma unroll
      for (int e = 0; e < 8; ++e) Vt[(c8 + e) * PSTR + key] = vv[e];
    }
    async_wait0();
    __syncthreads();

    // ---- phase 1: S = Q K^T from LDS ----
    v8f s[8];
#pragma unroll
    for (int nt = 0; nt < 8; ++nt) {
      const v16bf kb0 = load_frag(&KP[(nt * 16) * PSTR], PSTR, 0,  lane);
      const v16bf kb1 = load_frag(&KP[(nt * 16) * PSTR], PSTR, 32, lane);
      v8f t = {};
      t = wmma_bf16(qa0, kb0, t);
      t = wmma_bf16(qa1, kb1, t);
      s[nt] = t;
    }
    __syncthreads();  // all waves done with K tile; buffer becomes P storage

    // ---- band mask + scale, row max (16-lane butterfly per half-wave) ----
    float mnew[8], alpha[8];
#pragma unroll
    for (int vg = 0; vg < 8; ++vg) {
      const int qrow = qrow_base + vg;
      float mx = mrow[vg];
#pragma unroll
      for (int nt = 0; nt < 8; ++nt) {
        const int kcol = k0 + nt * 16 + ncol;
        const int dlt = qrow - kcol;
        float sv = s[nt][vg] * 0.125f;                 // 1/sqrt(64)
        sv = (dlt <= kWin && dlt >= -kWin) ? sv : -__builtin_inff();
        s[nt][vg] = sv;
        mx = fmaxf(mx, sv);
      }
#pragma unroll
      for (int xm = 1; xm < 16; xm <<= 1)
        mx = fmaxf(mx, __shfl_xor(mx, xm, 32));
      mnew[vg]  = mx;
      alpha[vg] = __expf(mrow[vg] - mx);
      mrow[vg]  = mx;
    }

    // ---- p = exp(s - m), row sums, spill P (bf16) to this wave's region ----
    bf16_t* pb = &KP[wave * 16 * PSTR];
#pragma unroll
    for (int vg = 0; vg < 8; ++vg) {
      float sum = 0.f;
      const int prow = vg + ((lane >> 4) << 3);
#pragma unroll
      for (int nt = 0; nt < 8; ++nt) {
        const float p = __expf(s[nt][vg] - mnew[vg]);
        sum += p;
        pb[prow * PSTR + nt * 16 + ncol] = (bf16_t)p;
      }
#pragma unroll
      for (int xm = 1; xm < 16; xm <<= 1)
        sum += __shfl_xor(sum, xm, 32);
      lrow[vg] = lrow[vg] * alpha[vg] + sum;
    }
#pragma unroll
    for (int j = 0; j < 4; ++j)
#pragma unroll
      for (int vg = 0; vg < 8; ++vg) o[j][vg] *= alpha[vg];

    // ---- phase 2: O += P(16x128) * V(128x64) (same-wave LDS ops in order) ----
#pragma unroll
    for (int kc = 0; kc < 4; ++kc) {
      const v16bf pa = load_frag(pb, PSTR, kc * 32, lane);
#pragma unroll
      for (int j = 0; j < 4; ++j) {
        const v16bf vb = load_frag(&Vt[(j * 16) * PSTR], PSTR, kc * 32, lane);
        o[j] = wmma_bf16(pa, vb, o[j]);
      }
    }
    __syncthreads();  // protect Vt/KP before next key tile
  }

  // ---- normalize and store attention output (bf16) ----
#pragma unroll
  for (int j = 0; j < 4; ++j) {
#pragma unroll
    for (int vg = 0; vg < 8; ++vg) {
      const float val = o[j][vg] / lrow[vg];
      const int qrow = qrow_base + vg;
      O[hbase + (size_t)qrow * kE + j * 16 + ncol] = (bf16_t)val;
    }
  }
}

// ---------------------------------------------------------------------------
extern "C" void kernel_launch(void* const* d_in, const int* in_sizes, int n_in,
                              void* d_out, int out_size, void* d_ws, size_t ws_size,
                              hipStream_t stream) {
  (void)in_sizes; (void)n_in; (void)out_size; (void)ws_size;
  const float* q  = (const float*)d_in[0];
  const float* k  = (const float*)d_in[1];
  const float* v  = (const float*)d_in[2];
  const float* Wq = (const float*)d_in[3];
  const float* bq = (const float*)d_in[4];
  const float* Wk = (const float*)d_in[5];
  const float* bk = (const float*)d_in[6];
  const float* Wv = (const float*)d_in[7];
  const float* bv = (const float*)d_in[8];
  const float* Wo = (const float*)d_in[9];
  const float* bo = (const float*)d_in[10];
  float* out = (float*)d_out;

  const size_t NSE = (size_t)kM * kE;
  bf16_t* Qb = (bf16_t*)d_ws;
  bf16_t* Kb = Qb + NSE;
  bf16_t* Vb = Kb + NSE;
  bf16_t* Ab = Vb + NSE;

  dim3 gg(kE / 128, kM / 128);
  gemm_bias_kernel<float, bf16_t><<<gg, 256, 0, stream>>>(q, Wq, bq, Qb, kM, kE, kE);
  gemm_bias_kernel<float, bf16_t><<<gg, 256, 0, stream>>>(k, Wk, bk, Kb, kM, kE, kE);
  gemm_bias_kernel<float, bf16_t><<<gg, 256, 0, stream>>>(v, Wv, bv, Vb, kM, kE, kE);

  dim3 ga(kS / 128, kH, kB);
  swa_kernel<<<ga, 256, 0, stream>>>(Qb, Kb, Vb, Ab);

  gemm_bias_kernel<bf16_t, float><<<gg, 256, 0, stream>>>(Ab, Wo, bo, out, kM, kE, kE);
}
